// ParallelRepeatHeads_76811195122559
// MI455X (gfx1250) — matrix operand order repaired and checked by reading, not compiled, not run on hardware
//
#include <hip/hip_runtime.h>
#include <hip/hip_bf16.h>

// Problem constants (B=2, S=2048, D=1024, H=16, HD=64)
#define PB 2
#define PS 2048
#define PD 1024
#define PH 16
#define PHD 64

typedef __bf16 bf16_t;
typedef __attribute__((ext_vector_type(16))) __bf16 v16bf;
typedef __attribute__((ext_vector_type(8)))  __bf16 v8bf;
typedef __attribute__((ext_vector_type(8)))  float  v8f;
typedef __attribute__((ext_vector_type(4)))  float  f32x4;

#if __has_builtin(__builtin_amdgcn_tensor_load_to_lds) && \
    __has_builtin(__builtin_amdgcn_s_wait_tensorcnt)
#define HAVE_TDM 1
typedef __attribute__((ext_vector_type(4))) unsigned int u32x4;
typedef __attribute__((ext_vector_type(8))) int          i32x8;
typedef __attribute__((ext_vector_type(4))) int          i32x4;
#endif

// The TDM writes LDS behind the compiler's back (descriptor carries only an
// integer LDS offset). Escape the buffer pointer through opaque asm with a
// memory clobber so the ds_load's of the staged tile cannot be undef-folded.
__device__ __forceinline__ void lds_written_by_tdm(void* p) {
    asm volatile("" : "+v"(p) :: "memory");
}

// ---------------- fragment loaders (layouts per CDNA5 ISA 7.12.2) -------------
// A fragment 16x32 bf16: lane = {ksel,mrow}; elements 0..7 = K kbase..kbase+7,
// elements 8..15 = K kbase+16..kbase+23, kbase = ksel*8.
// Works for global or LDS source pointers (address space inferred after inline).
__device__ __forceinline__ v16bf frag_a_f32(const float* row, int kbase) {
    f32x4 f0 = *(const f32x4*)(row + kbase);
    f32x4 f1 = *(const f32x4*)(row + kbase + 4);
    f32x4 f2 = *(const f32x4*)(row + kbase + 16);
    f32x4 f3 = *(const f32x4*)(row + kbase + 20);
    v16bf a;
#pragma unroll
    for (int i = 0; i < 4; ++i) {
        a[i]      = (__bf16)f0[i];
        a[4 + i]  = (__bf16)f1[i];
        a[8 + i]  = (__bf16)f2[i];
        a[12 + i] = (__bf16)f3[i];
    }
    return a;
}

// Same, with causal mask fused into the f32->bf16 convert: element j>t -> 0.
__device__ __forceinline__ v16bf frag_a_f32_causal(const float* row,
                                                   int kbase, int t, int j0) {
    f32x4 f0 = *(const f32x4*)(row + kbase);
    f32x4 f1 = *(const f32x4*)(row + kbase + 4);
    f32x4 f2 = *(const f32x4*)(row + kbase + 16);
    f32x4 f3 = *(const f32x4*)(row + kbase + 20);
    v16bf a;
#pragma unroll
    for (int i = 0; i < 4; ++i) {
        int jlo = j0 + kbase + i;
        int jhi = j0 + kbase + 16 + i;
        a[i]      = (jlo     <= t) ? (__bf16)f0[i] : (__bf16)0.0f;
        a[4 + i]  = (jlo + 4 <= t) ? (__bf16)f1[i] : (__bf16)0.0f;
        a[8 + i]  = (jhi     <= t) ? (__bf16)f2[i] : (__bf16)0.0f;
        a[12 + i] = (jhi + 4 <= t) ? (__bf16)f3[i] : (__bf16)0.0f;
    }
    return a;
}

// A fragment from bf16 source (two contiguous 16B loads).
__device__ __forceinline__ v16bf frag_a_bf16(const bf16_t* __restrict__ row, int kbase) {
    v8bf lo = *(const v8bf*)(row + kbase);
    v8bf hi = *(const v8bf*)(row + kbase + 16);
    v16bf a;
#pragma unroll
    for (int i = 0; i < 8; ++i) { a[i] = lo[i]; a[8 + i] = hi[i]; }
    return a;
}

// B fragment 32x16 bf16: lane = {ksel,ncol}; elements 0..7 = K koff..koff+7,
// elements 8..15 = K koff+16..koff+23, koff = ksel*8. Source row is the K-contiguous column (weight row / Xt row).
__device__ __forceinline__ v16bf frag_b_f32(const float* __restrict__ col, int koff) {
    f32x4 f0 = *(const f32x4*)(col + koff);
    f32x4 f1 = *(const f32x4*)(col + koff + 4);
    f32x4 f2 = *(const f32x4*)(col + koff + 16);
    f32x4 f3 = *(const f32x4*)(col + koff + 20);
    v16bf b;
#pragma unroll
    for (int i = 0; i < 4; ++i) {
        b[i]      = (__bf16)f0[i];
        b[4 + i]  = (__bf16)f1[i];
        b[8 + i]  = (__bf16)f2[i];
        b[12 + i] = (__bf16)f3[i];
    }
    return b;
}

__device__ __forceinline__ v16bf frag_b_bf16(const bf16_t* __restrict__ col, int koff) {
    v8bf lo = *(const v8bf*)(col + koff);
    v8bf hi = *(const v8bf*)(col + koff + 16);
    v16bf b;
#pragma unroll
    for (int i = 0; i < 8; ++i) { b[i] = lo[i]; b[8 + i] = hi[i]; }
    return b;
}

#define WMMA_BF16(ACC, A, B) \
    ACC = __builtin_amdgcn_wmma_f32_16x16x32_bf16(false, (A), false, (B), (short)0, (ACC), false, false)

// ============================ Kernel 1: in-projection =========================
// xp = x @ W_in^T + b_in ; store bf16 feature-major: Xt[(b*D+f)*S + s]
__global__ __launch_bounds__(256) void k_inproj(const float* __restrict__ x,
                                                const float* __restrict__ Win,
                                                const float* __restrict__ bin,
                                                bf16_t* __restrict__ xt) {
    const int lane = threadIdx.x & 31;
    const int wave = threadIdx.x >> 5;
    const int mrow = lane & 15, ksel = lane >> 4;
    const int kbA = ksel * 8, kbB = ksel * 8;
    const int m0 = blockIdx.y * 128 + (wave & 3) * 32;   // token rows
    const int n0 = blockIdx.x * 128 + (wave >> 2) * 64;  // output features

    v8f acc[2][4] = {};
    const float* arow[2];
    const float* brow[4];
#pragma unroll
    for (int mi = 0; mi < 2; ++mi) arow[mi] = x + (size_t)(m0 + mi * 16 + mrow) * PD;
#pragma unroll
    for (int ni = 0; ni < 4; ++ni) brow[ni] = Win + (size_t)(n0 + ni * 16 + mrow) * PD;

    for (int kb = 0; kb < PD; kb += 32) {
        v16bf a[2], b[4];
#pragma unroll
        for (int mi = 0; mi < 2; ++mi) a[mi] = frag_a_f32(arow[mi] + kb, kbA);
#pragma unroll
        for (int ni = 0; ni < 4; ++ni) b[ni] = frag_b_f32(brow[ni] + kb, kbB);
#pragma unroll
        for (int mi = 0; mi < 2; ++mi)
#pragma unroll
            for (int ni = 0; ni < 4; ++ni) WMMA_BF16(acc[mi][ni], a[mi], b[ni]);
    }

    // D layout: VGPR r -> row (ksel*8 + r), col = lane&15
    const int mrow8 = ksel * 8, ncol = lane & 15;
#pragma unroll
    for (int mi = 0; mi < 2; ++mi) {
        int mbase = m0 + mi * 16 + mrow8;
        int bb = mbase >> 11, s0 = mbase & (PS - 1);
#pragma unroll
        for (int ni = 0; ni < 4; ++ni) {
            int f = n0 + ni * 16 + ncol;
            float bias = bin[f];
            v8bf pack;
#pragma unroll
            for (int r = 0; r < 8; ++r) pack[r] = (__bf16)(acc[mi][ni][r] + bias);
            *(v8bf*)(xt + ((size_t)(bb * PD + f)) * PS + s0) = pack;  // 16B store
        }
    }
}

// ============================ Kernel 2: causal mixing =========================
// mixed[b,t,h,:] = sum_{j<=t} Wmix[h,t,j] * xh[b,j,h,:]
// A = Wmix[h] tile (128 rows x 32 K, fp32) staged to LDS by the Tensor Data
//     Mover (double buffered, TENSORcnt), TDM padding = 4 dwords per 32-dword
//     row -> LDS row stride 36 dwords => conflict-free ds_load_b128 fragments.
// B = Xt (bf16, K-contiguous), N = 128 = {b0 64 dims, b1 64 dims}
// out: token-major bf16 mixed[(b*S+t)*D + h*64+d]
#define AT_STRIDE 36                       // padded LDS row stride (dwords)
#define AT_DWORDS (128 * AT_STRIDE)        // one buffer, dwords
#define AT_BYTES  (AT_DWORDS * 4)

#if HAVE_TDM
// Issue one TDM 2D tile load: 128 rows x 32 fp32, row stride 2048 floats,
// LDS padding 4 dwords per 32-dword row. D# per CDNA5 ISA 8.3/8.4.
// This toolchain's builtin takes 6 args:
//   (u32x4 g0, i32x8 g1, i32x4 g2, i32x4 g3, i32x8, i32 cpol)
__device__ __forceinline__ void tdm_load_tile(unsigned lds_off, const float* gsrc) {
    unsigned long long ga = (unsigned long long)(uintptr_t)gsrc;
    u32x4 g0;
    g0[0] = 1u;                                   // count=1 (valid, user mode)
    g0[1] = lds_off;                              // lds_addr (bytes)
    g0[2] = (unsigned)(ga & 0xFFFFFFFFu);         // global_addr[31:0]
    g0[3] = (unsigned)((ga >> 32) & 0x01FFFFFFu)  // global_addr[56:32]
          | (2u << 30);                           // type = 2 ("image")
    const unsigned td0 = 1u << 20, td1 = 1u << 20;  // tensor dims (no OOB)
    i32x8 g1;
    g1[0] = (2 << 16)        // data_size = 4 bytes
          | (1 << 20)        // pad_enable
          | (4 << 22)        // pad_interval: 32 dwords
          | (3 << 25);       // pad_amount: 4 dwords
    g1[1] = (int)((td0 & 0xFFFFu) << 16);                       // tensor_dim0 lo
    g1[2] = (int)(((td0 >> 16) & 0xFFFFu) | ((td1 & 0xFFFFu) << 16));
    g1[3] = (int)(((td1 >> 16) & 0xFFFFu) | (32u << 16));       // tile_dim0 = 32
    g1[4] = 128;                                                // tile_dim1 = 128
    g1[5] = 2048;                                               // dim0_stride lo
    g1[6] = 0;
    g1[7] = 0;
    i32x4 z4 = {0, 0, 0, 0};             // groups 2/3 unused (2D tensor)
    i32x8 z8 = {0, 0, 0, 0, 0, 0, 0, 0};
    __builtin_amdgcn_tensor_load_to_lds(g0, g1, z4, z4, z8, 0);
}
#endif

__global__ __launch_bounds__(256) void k_mix(const float* __restrict__ Wmix,
                                             const bf16_t* __restrict__ xt,
                                             bf16_t* __restrict__ mixed) {
    __shared__ float atile[2][AT_DWORDS];

    const int h = blockIdx.y;
    const int lane = threadIdx.x & 31;
    const int wave = threadIdx.x >> 5;
    const int mrow = lane & 15, ksel = lane >> 4;
    const int kbA = ksel * 8, kbB = ksel * 8;
    const int wm = wave & 3, wn = wave >> 2;
    const int t0_blk = blockIdx.x * 128;
    const int t0w = t0_blk + wm * 32;   // this wave's first output row t
    const int n0 = wn * 64;             // cols within 128

    v8f acc[2][4] = {};
    int trow[2];
#pragma unroll
    for (int mi = 0; mi < 2; ++mi) trow[mi] = t0w + mi * 16 + mrow;
    const bf16_t* brow[4];
#pragma unroll
    for (int ni = 0; ni < 4; ++ni) {
        int n = n0 + ni * 16 + mrow;
        int bb = n >> 6, d = n & 63;
        brow[ni] = xt + ((size_t)(bb * PD + h * PHD + d)) * PS;
    }

    const float* gbase = Wmix + ((size_t)h * PS + t0_blk) * PS;  // block A rows
    const int wtmax = t0w + 31;                 // last row this wave owns
    const int nk = 4 * (blockIdx.x + 1);        // K-tiles up to the diagonal

#if HAVE_TDM
    if (wave == 0) tdm_load_tile(0u, gbase);    // prologue: fill buffer 0
#endif

    for (int jt = 0; jt < nk; ++jt) {
        const int j0 = jt << 5;
#if HAVE_TDM
        if (wave == 0) {
            if (jt + 1 < nk) {
                tdm_load_tile((unsigned)(((jt + 1) & 1) * AT_BYTES),
                              gbase + (size_t)(j0 + 32));
                __builtin_amdgcn_s_wait_tensorcnt(1);  // current buffer ready
            } else {
                __builtin_amdgcn_s_wait_tensorcnt(0);
            }
        }
        __syncthreads();                        // buffer visible to all waves
        lds_written_by_tdm(&atile[0][0]);       // compiler: TDM wrote this LDS
        const int buf = jt & 1;
#else
        {   // cooperative staging fallback: 256 thr x 4 b128 = 128x32 fp32
            const int r  = threadIdx.x >> 1;
            const int c0 = (threadIdx.x & 1) * 16;
            const float* g = gbase + (size_t)r * PS + j0 + c0;
#pragma unroll
            for (int c = 0; c < 4; ++c)
                *(f32x4*)&atile[0][r * AT_STRIDE + c0 + c * 4] =
                    *(const f32x4*)(g + c * 4);
        }
        __syncthreads();
        const int buf = 0;
#endif
        if (j0 <= wtmax) {                      // skip tiles above the diagonal
            v16bf a[2], b[4];
            const bool maskme = (j0 + 31 > t0w);
#pragma unroll
            for (int mi = 0; mi < 2; ++mi) {
                const float* rowp = &atile[buf][(wm * 32 + mi * 16 + mrow) * AT_STRIDE];
                a[mi] = maskme ? frag_a_f32_causal(rowp, kbA, trow[mi], j0)
                               : frag_a_f32(rowp, kbA);
            }
#pragma unroll
            for (int ni = 0; ni < 4; ++ni) b[ni] = frag_b_bf16(brow[ni] + j0, kbB);
#pragma unroll
            for (int mi = 0; mi < 2; ++mi)
#pragma unroll
                for (int ni = 0; ni < 4; ++ni) WMMA_BF16(acc[mi][ni], a[mi], b[ni]);
        }
        __syncthreads();                        // compute done before overwrite
    }

    const int mrow8 = ksel * 8, ncol = lane & 15;
#pragma unroll
    for (int mi = 0; mi < 2; ++mi)
#pragma unroll
        for (int ni = 0; ni < 4; ++ni) {
            int n = n0 + ni * 16 + ncol;
            int bb = n >> 6, f = h * PHD + (n & 63);
#pragma unroll
            for (int r = 0; r < 8; ++r) {
                int t = t0w + mi * 16 + mrow8 + r;
                mixed[((size_t)(bb * PS + t)) * PD + f] = (__bf16)acc[mi][ni][r];
            }
        }
}

// ============================ Kernel 3: out-projection ========================
// out = mixed @ W_out^T + b_out (fp32 output)
__global__ __launch_bounds__(256) void k_outproj(const bf16_t* __restrict__ mixed,
                                                 const float* __restrict__ Wout,
                                                 const float* __restrict__ bout,
                                                 float* __restrict__ out) {
    const int lane = threadIdx.x & 31;
    const int wave = threadIdx.x >> 5;
    const int mrow = lane & 15, ksel = lane >> 4;
    const int kbA = ksel * 8, kbB = ksel * 8;
    const int m0 = blockIdx.y * 128 + (wave & 3) * 32;
    const int n0 = blockIdx.x * 128 + (wave >> 2) * 64;

    v8f acc[2][4] = {};
    const bf16_t* arow[2];
    const float* brow[4];
#pragma unroll
    for (int mi = 0; mi < 2; ++mi) arow[mi] = mixed + (size_t)(m0 + mi * 16 + mrow) * PD;
#pragma unroll
    for (int ni = 0; ni < 4; ++ni) brow[ni] = Wout + (size_t)(n0 + ni * 16 + mrow) * PD;

    for (int kb = 0; kb < PD; kb += 32) {
        v16bf a[2], b[4];
#pragma unroll
        for (int mi = 0; mi < 2; ++mi) a[mi] = frag_a_bf16(arow[mi] + kb, kbA);
#pragma unroll
        for (int ni = 0; ni < 4; ++ni) b[ni] = frag_b_f32(brow[ni] + kb, kbB);
#pragma unroll
        for (int mi = 0; mi < 2; ++mi)
#pragma unroll
            for (int ni = 0; ni < 4; ++ni) WMMA_BF16(acc[mi][ni], a[mi], b[ni]);
    }

    const int mrow8 = ksel * 8, ncol = lane & 15;
#pragma unroll
    for (int mi = 0; mi < 2; ++mi)
#pragma unroll
        for (int ni = 0; ni < 4; ++ni) {
            int n = n0 + ni * 16 + ncol;
            float bias = bout[n];
#pragma unroll
            for (int r = 0; r < 8; ++r) {
                int m = m0 + mi * 16 + mrow8 + r;
                out[(size_t)m * PD + n] = acc[mi][ni][r] + bias;
            }
        }
}

// ================================ launcher ====================================
extern "C" void kernel_launch(void* const* d_in, const int* in_sizes, int n_in,
                              void* d_out, int out_size, void* d_ws, size_t ws_size,
                              hipStream_t stream) {
    const float* x    = (const float*)d_in[0];
    const float* Win  = (const float*)d_in[1];
    const float* bin  = (const float*)d_in[2];
    const float* Wmix = (const float*)d_in[3];
    const float* Wout = (const float*)d_in[4];
    const float* bout = (const float*)d_in[5];
    float* out = (float*)d_out;

    bf16_t* xt    = (bf16_t*)d_ws;                                   // 8 MB
    bf16_t* mixed = (bf16_t*)((char*)d_ws + (size_t)PB * PD * PS * sizeof(bf16_t));

    dim3 blk(256);
    // in-projection: M=4096 tokens, N=1024 features, 128x128 block tiles
    k_inproj<<<dim3(PD / 128, (PB * PS) / 128), blk, 0, stream>>>(x, Win, bin, xt);
    // mixing: one block per (t-block, head); N=128 covers both batches
    k_mix<<<dim3(PS / 128, PH), blk, 0, stream>>>(Wmix, xt, mixed);
    // out-projection
    k_outproj<<<dim3(PD / 128, (PB * PS) / 128), blk, 0, stream>>>(mixed, Wout, bout, out);
}
